// TransformationGAT_2095944040974
// MI455X (gfx1250) — matrix operand order repaired and checked by reading, not compiled, not run on hardware
//
#include <hip/hip_runtime.h>
#include <hip/hip_bf16.h>
#include <math.h>

typedef __attribute__((ext_vector_type(16))) _Float16 v16h;
typedef __attribute__((ext_vector_type(8)))  _Float16 v8h;
typedef __attribute__((ext_vector_type(8)))  float    v8f;

#define H_HEADS 4
#define C_CH    32
#define HC      128      // H_HEADS * C_CH
#define TD      256
#define DEC     512
#define NEG_SLOPE 0.2f

#define SW_STRIDE 136    // padded LDS column stride (f16 elems) -> conflict-free b128 reads

// ---- order-preserving float<->int mapping for atomicMax on signed int ----
__device__ __forceinline__ int enc_f32(float f) {
    int i = __float_as_int(f);
    return (i >= 0) ? i : (i ^ 0x7FFFFFFF);
}
__device__ __forceinline__ float dec_f32(int i) {
    int j = (i >= 0) ? i : (i ^ 0x7FFFFFFF);
    return __int_as_float(j);
}

// ---------------- generic fills (graph-capture-safe zeroing) ----------------
__global__ void fill_f32(float* __restrict__ p, float v, long long n) {
    long long i = (long long)blockIdx.x * blockDim.x + threadIdx.x;
    if (i < n) p[i] = v;
}
__global__ void fill_i32(int* __restrict__ p, int v, long long n) {
    long long i = (long long)blockIdx.x * blockDim.x + threadIdx.x;
    if (i < n) p[i] = v;
}

// ---------------- input concat + fp32 -> fp16 ----------------
__global__ void concat_to_f16(const float* __restrict__ enz, const float* __restrict__ sub,
                              _Float16* __restrict__ x, int N) {
    long long i = (long long)blockIdx.x * blockDim.x + threadIdx.x;
    if (i >= (long long)N * HC) return;
    int n = (int)(i >> 7);
    int k = (int)(i & 127);
    float v = (k < 64) ? enz[(long long)n * 64 + k] : sub[(long long)n * 64 + (k - 64)];
    x[i] = (_Float16)v;
}

// ---------------- weight [K=128][N=128] row-major fp32 -> col-major fp16 ----------------
__global__ void w_to_f16_T(const float* __restrict__ W, _Float16* __restrict__ Wt) {
    int i = blockIdx.x * blockDim.x + threadIdx.x;   // 16384 threads
    if (i >= HC * HC) return;
    int n = i >> 7, k = i & 127;
    // Wt[n*128 + k] = W[k*128 + n]
    Wt[i] = (_Float16)W[k * HC + n];
}

// ---------------- WMMA GEMM: H = X[N,128] @ W[128,128] ----------------
// 256-thread block = 8 waves. Block stages Wt (col-major) into padded LDS once;
// each wave computes 32 rows (two 16-row tiles) x 128 cols, so every LDS B
// fragment feeds two back-to-back v_wmma ops.
__global__ __launch_bounds__(256)
void gemm32(const _Float16* __restrict__ X, const _Float16* __restrict__ Wt,
            float* __restrict__ Hh, int nwavetiles) {
    __shared__ _Float16 sW[HC * SW_STRIDE];     // 34 KB of the 320 KB WGP LDS
    const int t = threadIdx.x;

    // ---- cooperative stage: 2 threads per column, 64 f16 (8 x v8h) each ----
    {
        const int n  = t >> 1;          // column 0..127
        const int hk = t & 1;           // K half 0..1
        const v8h* src = reinterpret_cast<const v8h*>(Wt + n * HC + hk * 64);
        v8h*       dst = reinterpret_cast<v8h*>(sW + n * SW_STRIDE + hk * 64);
#pragma unroll
        for (int j = 0; j < 8; ++j) dst[j] = src[j];
    }
    __syncthreads();

    const int wave = t >> 5;
    const int lane = t & 31;
    const int m    = lane & 15;         // A row within tile / B,C,D column within tile
    const int half = lane >> 4;         // half-wave select

    const long long wt = (long long)blockIdx.x * 8 + wave;   // 32-row wave-tile index
    if (wt >= nwavetiles) return;                            // wave-uniform guard
    const long long row0 = wt * 32;

    // ---- A fragments for two 16-row tiles (ISA 16-bit 16x32 layout) ----
    // elems 0..7 -> K = 32*kk + half*8 + e ; elems 8..15 -> K = 32*kk + 16 + half*8 + (e-8)
    v16h a0[4], a1[4];
    const _Float16* xr0 = X + (row0 + m) * HC;
    const _Float16* xr1 = X + (row0 + 16 + m) * HC;
#pragma unroll
    for (int kk = 0; kk < 4; ++kk) {
        const v8h lo0 = *reinterpret_cast<const v8h*>(xr0 + kk * 32 + half * 8);
        const v8h hi0 = *reinterpret_cast<const v8h*>(xr0 + kk * 32 + half * 8 + 16);
        const v8h lo1 = *reinterpret_cast<const v8h*>(xr1 + kk * 32 + half * 8);
        const v8h hi1 = *reinterpret_cast<const v8h*>(xr1 + kk * 32 + half * 8 + 16);
        v16h av0, av1;
#pragma unroll
        for (int i = 0; i < 8; ++i) {
            av0[i] = lo0[i]; av0[i + 8] = hi0[i];
            av1[i] = lo1[i]; av1[i + 8] = hi1[i];
        }
        a0[kk] = av0; a1[kk] = av1;
    }

#pragma unroll
    for (int ct = 0; ct < 8; ++ct) {                 // 8 column tiles of 16
        v8f c0 = {}, c1 = {};
#pragma unroll
        for (int kk = 0; kk < 4; ++kk) {
            // B 32x16: lane holds column n = ct*16+m; elems 0..7 -> K = 32*kk + half*8 + e ; elems 8..15 -> K = 32*kk + 16 + half*8 + (e-8)
            const _Float16* bp = sW + (ct * 16 + m) * SW_STRIDE + kk * 32 + half * 8;
            const v8h blo = *reinterpret_cast<const v8h*>(bp);
            const v8h bhi = *reinterpret_cast<const v8h*>(bp + 16);
            v16h b;
#pragma unroll
            for (int i = 0; i < 8; ++i) { b[i] = blo[i]; b[i + 8] = bhi[i]; }
            c0 = __builtin_amdgcn_wmma_f32_16x16x32_f16(false, a0[kk], false, b, (short)0, c0, false, false);
            c1 = __builtin_amdgcn_wmma_f32_16x16x32_f16(false, a1[kk], false, b, (short)0, c1, false, false);
        }
        // C/D: VGPR r, lanes 0-15: M=r ; lanes 16-31: M=r+8 ; N = m
#pragma unroll
        for (int r = 0; r < 8; ++r) {
            Hh[(row0 + half * 8 + r) * HC + ct * 16 + m]      = c0[r];
            Hh[(row0 + 16 + half * 8 + r) * HC + ct * 16 + m] = c1[r];
        }
    }
}

// ---------------- per-node attention logits alpha_src / alpha_dst ----------------
__global__ void alphas_k(const float* __restrict__ Hh, const float* __restrict__ asrc,
                         const float* __restrict__ adst, float* __restrict__ as_,
                         float* __restrict__ ad_, int N) {
    long long i = (long long)blockIdx.x * blockDim.x + threadIdx.x;
    if (i >= (long long)N * H_HEADS) return;
    int n  = (int)(i >> 2);
    int hd = (int)(i & 3);
    const float* hp = Hh + (long long)n * HC + hd * C_CH;
    const float* sp = asrc + hd * C_CH;
    const float* dp = adst + hd * C_CH;
    float s = 0.f, d = 0.f;
#pragma unroll
    for (int c = 0; c < C_CH; ++c) { float hv = hp[c]; s += hv * sp[c]; d += hv * dp[c]; }
    as_[i] = s;
    ad_[i] = d;
}

// ---------------- edge pass 1: leaky-relu logits + segment max ----------------
__global__ void edge_logit(const int* __restrict__ ei, int E, int N,
                           const float* __restrict__ as_, const float* __restrict__ ad_,
                           float* __restrict__ eval, int* __restrict__ mmax) {
    long long i = (long long)blockIdx.x * blockDim.x + threadIdx.x;
    long long EP = (long long)E + N;
    if (i >= EP * H_HEADS) return;
    long long e = i >> 2;
    int hd = (int)(i & 3);
    int src = (e < E) ? ei[e]     : (int)(e - E);
    int dst = (e < E) ? ei[E + e] : (int)(e - E);
    float v = as_[(long long)src * H_HEADS + hd] + ad_[(long long)dst * H_HEADS + hd];
    v = (v > 0.f) ? v : NEG_SLOPE * v;
    eval[i] = v;
    atomicMax(&mmax[(long long)dst * H_HEADS + hd], enc_f32(v));
}

// ---------------- edge pass 2: exp(v - max) + segment sum ----------------
__global__ void edge_exp(const int* __restrict__ ei, int E, int N,
                         const int* __restrict__ mmax, float* __restrict__ eval,
                         float* __restrict__ z) {
    long long i = (long long)blockIdx.x * blockDim.x + threadIdx.x;
    long long EP = (long long)E + N;
    if (i >= EP * H_HEADS) return;
    long long e = i >> 2;
    int hd = (int)(i & 3);
    int dst = (e < E) ? ei[E + e] : (int)(e - E);
    float mx = dec_f32(mmax[(long long)dst * H_HEADS + hd]);
    float ex = expf(eval[i] - mx);
    eval[i] = ex;
    atomicAdd(&z[(long long)dst * H_HEADS + hd], ex);
}

// ---------------- edge pass 3: weighted message scatter ----------------
__global__ void edge_aggr(const int* __restrict__ ei, int E, int N,
                          const float* __restrict__ eval, const float* __restrict__ z,
                          const float* __restrict__ Hh, float* __restrict__ acc) {
    long long i = (long long)blockIdx.x * blockDim.x + threadIdx.x;
    long long EP = (long long)E + N;
    if (i >= EP * HC) return;
    long long e = i >> 7;
    int c  = (int)(i & 127);
    int hd = c >> 5;
    int src = (e < E) ? ei[e]     : (int)(e - E);
    int dst = (e < E) ? ei[E + e] : (int)(e - E);
    float a = eval[e * H_HEADS + hd] / z[(long long)dst * H_HEADS + hd];
    atomicAdd(&acc[(long long)dst * HC + c], Hh[(long long)src * HC + c] * a);
}

// ---------------- bias + ELU; optionally emit f16 (next GEMM) and/or f32 (mean) ----------------
__global__ void bias_elu(const float* __restrict__ acc, const float* __restrict__ b,
                         _Float16* __restrict__ x16, float* __restrict__ x32,
                         int N, int w16, int w32) {
    long long i = (long long)blockIdx.x * blockDim.x + threadIdx.x;
    if (i >= (long long)N * HC) return;
    int c = (int)(i & 127);
    float v = acc[i] + b[c];
    float el = (v > 0.f) ? v : (expf(v) - 1.f);
    if (w16) x16[i] = (_Float16)el;
    if (w32) x32[i] = el;
}

// ---------------- mean over nodes: 128 channels, block handles 128-node chunk ----------------
__global__ __launch_bounds__(128)
void mean_reduce(const float* __restrict__ x, float* __restrict__ g, int N) {
    int c  = threadIdx.x;
    int n0 = blockIdx.x * 128;
    int n1 = (n0 + 128 < N) ? n0 + 128 : N;
    float s = 0.f;
    for (int n = n0; n < n1; ++n) s += x[(long long)n * HC + c];
    atomicAdd(&g[c], s);
}

// ---------------- fc: pred = (g/N) @ Wfc + bfc ----------------
__global__ __launch_bounds__(256)
void fc_pred(const float* __restrict__ g, const float* __restrict__ Wfc,
             const float* __restrict__ bfc, float* __restrict__ pred, float invN) {
    int t = threadIdx.x;                 // 0..255
    float s = bfc[t];
    for (int k = 0; k < HC; ++k) s += (g[k] * invN) * Wfc[k * TD + t];
    pred[t] = s;
}

// ---------------- out = pred @ Wdec + bdec ----------------
__global__ __launch_bounds__(256)
void fc_out(const float* __restrict__ pred, const float* __restrict__ Wdec,
            const float* __restrict__ bdec, float* __restrict__ out) {
    int j = blockIdx.x * 256 + threadIdx.x;   // 0..511
    if (j >= DEC) return;
    float s = bdec[j];
    for (int t = 0; t < TD; ++t) s += pred[t] * Wdec[t * DEC + j];
    out[j] = s;
}

// ============================================================================
extern "C" void kernel_launch(void* const* d_in, const int* in_sizes, int n_in,
                              void* d_out, int out_size, void* d_ws, size_t ws_size,
                              hipStream_t stream) {
    const float* enz  = (const float*)d_in[0];
    const float* sub  = (const float*)d_in[1];
    const int*   ei   = (const int*)  d_in[2];
    const float* W1   = (const float*)d_in[3];
    const float* aS1  = (const float*)d_in[4];
    const float* aD1  = (const float*)d_in[5];
    const float* b1   = (const float*)d_in[6];
    const float* W2   = (const float*)d_in[7];
    const float* aS2  = (const float*)d_in[8];
    const float* aD2  = (const float*)d_in[9];
    const float* b2   = (const float*)d_in[10];
    const float* Wfc  = (const float*)d_in[11];
    const float* bfc  = (const float*)d_in[12];
    const float* Wdec = (const float*)d_in[13];
    const float* bdec = (const float*)d_in[14];

    const int N  = in_sizes[0] / 64;
    const int E  = in_sizes[2] / 2;
    const long long EP = (long long)E + N;

    // ---- workspace carve-up (256B aligned) ----
    char* ws = (char*)d_ws;
    size_t off = 0;
    auto alloc = [&](size_t bytes) { char* p = ws + off; off = (off + bytes + 255) & ~(size_t)255; return p; };
    _Float16* xf16 = (_Float16*)alloc((size_t)N * HC * sizeof(_Float16));
    float*    hbuf = (float*)   alloc((size_t)N * HC * sizeof(float));    // GEMM out; reused as post-ELU f32
    float*    accb = (float*)   alloc((size_t)N * HC * sizeof(float));
    float*    evalb= (float*)   alloc((size_t)EP * H_HEADS * sizeof(float));
    float*    as_  = (float*)   alloc((size_t)N * H_HEADS * sizeof(float));
    float*    ad_  = (float*)   alloc((size_t)N * H_HEADS * sizeof(float));
    int*      mbuf = (int*)     alloc((size_t)N * H_HEADS * sizeof(int));
    float*    zbuf = (float*)   alloc((size_t)N * H_HEADS * sizeof(float));
    _Float16* Wt1  = (_Float16*)alloc((size_t)HC * HC * sizeof(_Float16));
    _Float16* Wt2  = (_Float16*)alloc((size_t)HC * HC * sizeof(_Float16));
    float*    gbuf = (float*)   alloc(HC * sizeof(float));
    float*    pred = (float*)   alloc(TD * sizeof(float));
    (void)ws_size; (void)n_in; (void)out_size;

    const long long nNH  = (long long)N * HC;          // N*128
    const long long nEH  = EP * H_HEADS;               // E'*4
    const long long nEC  = EP * HC;                    // E'*128
    auto blks = [](long long n, int t) { return (unsigned int)((n + t - 1) / t); };

    const int NEG_INF_ENC = (int)0x807FFFFF;           // enc_f32(-inf)
    const int nwavetiles  = (N + 31) / 32;             // 32-row tiles (N=100000 -> 3125 exact)
    const unsigned gemm_blocks = (unsigned)((nwavetiles + 7) / 8);

    // ---- prep ----
    concat_to_f16<<<blks(nNH, 256), 256, 0, stream>>>(enz, sub, xf16, N);
    w_to_f16_T  <<<64, 256, 0, stream>>>(W1, Wt1);
    w_to_f16_T  <<<64, 256, 0, stream>>>(W2, Wt2);

    // ---- two GAT layers ----
    for (int layer = 0; layer < 2; ++layer) {
        const _Float16* Wt = layer == 0 ? Wt1 : Wt2;
        const float* asrc  = layer == 0 ? aS1 : aS2;
        const float* adst  = layer == 0 ? aD1 : aD2;
        const float* bias  = layer == 0 ? b1  : b2;

        gemm32<<<gemm_blocks, 256, 0, stream>>>(xf16, Wt, hbuf, nwavetiles);
        alphas_k<<<blks((long long)N * H_HEADS, 256), 256, 0, stream>>>(hbuf, asrc, adst, as_, ad_, N);

        fill_i32<<<blks((long long)N * H_HEADS, 256), 256, 0, stream>>>(mbuf, NEG_INF_ENC, (long long)N * H_HEADS);
        fill_f32<<<blks((long long)N * H_HEADS, 256), 256, 0, stream>>>(zbuf, 0.f, (long long)N * H_HEADS);
        fill_f32<<<blks(nNH, 256), 256, 0, stream>>>(accb, 0.f, nNH);

        edge_logit<<<blks(nEH, 256), 256, 0, stream>>>(ei, E, N, as_, ad_, evalb, mbuf);
        edge_exp  <<<blks(nEH, 256), 256, 0, stream>>>(ei, E, N, mbuf, evalb, zbuf);
        edge_aggr <<<blks(nEC, 256), 256, 0, stream>>>(ei, E, N, evalb, zbuf, hbuf, accb);

        // layer 0: produce f16 input for next GEMM; layer 1: produce f32 for the mean (into hbuf)
        bias_elu<<<blks(nNH, 256), 256, 0, stream>>>(accb, bias, xf16, hbuf, N,
                                                     layer == 0 ? 1 : 0, layer == 1 ? 1 : 0);
    }

    // ---- mean pool + FC head ----
    fill_f32<<<1, HC, 0, stream>>>(gbuf, 0.f, HC);
    mean_reduce<<<blks(N, 128), 128, 0, stream>>>(hbuf, gbuf, N);
    fc_pred<<<1, 256, 0, stream>>>(gbuf, Wfc, bfc, pred, 1.f / (float)N);
    fc_out<<<2, 256, 0, stream>>>(pred, Wdec, bdec, (float*)d_out);
}
